// SeparateHiddenPRADADecoder_369367188155
// MI455X (gfx1250) — matrix operand order repaired and checked by reading, not compiled, not run on hardware
//
#include <hip/hip_runtime.h>
#include <hip/hip_bf16.h>

typedef __attribute__((ext_vector_type(16))) __bf16 v16bf;
typedef __attribute__((ext_vector_type(8)))  __bf16 v8bf;
typedef __attribute__((ext_vector_type(8)))  float  v8f;

#define N_NODES 100000
#define N_EDGES 3200000
#define F_IN    128
#define F_CAT   256
#define F_HID   256
#define F_H2    512
#define F_OUT   128
#define ROW_TILES (N_NODES/16)   // 6250, exact

// workspace layout (bytes)
#define OFF_DIS   0ull
#define OFF_XS    (1ull<<20)
#define SZ_BIG    ((size_t)N_NODES * F_CAT * 4ull)       // 102,400,000 B
#define OFF_AGG   (OFF_XS + SZ_BIG)
#define OFF_PACK  (OFF_AGG + SZ_BIG)

static __device__ __forceinline__ v16bf assemble16(v8bf lo, v8bf hi) {
    v16bf r;
#pragma unroll
    for (int i = 0; i < 8; ++i) { r[i] = lo[i]; r[8 + i] = hi[i]; }
    return r;
}

// ---------------- degree / norm ----------------
__global__ void k_init_deg(float* deg) {
    int n = blockIdx.x * blockDim.x + threadIdx.x;
    if (n < N_NODES) deg[n] = 1.0f;   // self loop
}

__global__ void k_count_deg(const long long* ei, float* deg) {
    int e = blockIdx.x * blockDim.x + threadIdx.x;
    if (e < N_EDGES) {
        int dst = (int)ei[N_EDGES + e];
        unsafeAtomicAdd(&deg[dst], 1.0f);
    }
}

__global__ void k_dis(float* deg) {
    int n = blockIdx.x * blockDim.x + threadIdx.x;
    if (n < N_NODES) deg[n] = rsqrtf(deg[n]);   // deg >= 1 always
}

// ------------- pre-scale + self-loop init -------------
__global__ void k_scale_init(const float* __restrict__ latent,
                             const float* __restrict__ cond,
                             const float* __restrict__ dis,
                             float4* __restrict__ xs, float4* __restrict__ agg) {
    int tid = blockIdx.x * blockDim.x + threadIdx.x;   // N*64 threads
    int n = tid >> 6, c4 = tid & 63;
    if (n >= N_NODES) return;
    float d = dis[n];
    float4 v = (c4 < 32) ? ((const float4*)latent)[(size_t)n*32 + c4]
                         : ((const float4*)cond)[(size_t)n*32 + (c4-32)];
    v.x *= d; v.y *= d; v.z *= d; v.w *= d;
    size_t idx = (size_t)n*64 + c4;
    xs[idx] = v;
    agg[idx] = v;
}

// ------------- edge scatter, stage A (256 f32 / edge) -------------
__global__ void k_scatter1(const long long* __restrict__ ei,
                           const float* __restrict__ xs, float* __restrict__ agg) {
    int tid = blockIdx.x * blockDim.x + threadIdx.x;   // E*64 threads
    int e = tid >> 6, c = (tid & 63) << 2;
    if (e >= N_EDGES) return;
    int src = (int)ei[e];
    int dst = (int)ei[N_EDGES + e];
    float4 v = *(const float4*)(xs + (size_t)src * F_CAT + c);
    float* o = agg + (size_t)dst * F_CAT + c;
    unsafeAtomicAdd(o + 0, v.x);
    unsafeAtomicAdd(o + 1, v.y);
    unsafeAtomicAdd(o + 2, v.z);
    unsafeAtomicAdd(o + 3, v.w);
}

// ------------- pack fp32 row-major W[K,Ncols] into wave32 bf16 B panels -------------
// lane L of (ktile kt, ntile nt): col = nt*16 + (L&15), K = kt*32 + (L<16?0:8) + i + (i<8?0:8)
__global__ void k_pack(const float* __restrict__ W, __bf16* __restrict__ P,
                       int KT, int NT, int ldn) {
    int tid = blockIdx.x * blockDim.x + threadIdx.x;
    if (tid >= KT * NT * 32) return;
    int kt   = tid / (NT * 32);
    int rem  = tid - kt * NT * 32;
    int nt   = rem >> 5;
    int lane = rem & 31;
    int col  = nt * 16 + (lane & 15);
    int kb   = kt * 32 + ((lane & 16) ? 8 : 0);
    __bf16* p = P + ((size_t)(kt * NT + nt) * 32 + lane) * 16;
#pragma unroll
    for (int i = 0; i < 16; ++i)
        p[i] = (__bf16)W[(size_t)(kb + i + (i >= 8 ? 8 : 0)) * ldn + col];
}

// ------------- GEMM1: h = tanh(dis ⊙ (agg @ blockdiag(Wz,Wc)) + bias), bf16 out ----
// One block per 16-row tile. A (16x256, converted to bf16 A-fragment layout) is staged
// once in LDS and shared by all 8 waves. Each wave: 4 column tiles, 4 accumulators.
__global__ __launch_bounds__(256) void k_gemm1(const float* __restrict__ agg,
                                               const float* __restrict__ dis,
                                               const __bf16* __restrict__ Pz,
                                               const __bf16* __restrict__ Pc,
                                               const float* __restrict__ bz,
                                               const float* __restrict__ bc,
                                               __bf16* __restrict__ h) {
    __shared__ __bf16 As[8 * 32 * 16];   // 8 k-tiles x 32 lanes x 16 halves = 8 KB
    const int t = threadIdx.x;
    const int r0 = blockIdx.x * 16;
    {   // cooperative A staging: thread = (kt, lane)
        int kt = t >> 5, lane = t & 31;
        int m = lane & 15;
        int kb = kt * 32 + ((lane & 16) ? 8 : 0);
        const float* arow = agg + (size_t)(r0 + m) * F_CAT;
        float4 a0 = *(const float4*)(arow + kb);
        float4 a1 = *(const float4*)(arow + kb + 4);
        float4 a2 = *(const float4*)(arow + kb + 16);
        float4 a3 = *(const float4*)(arow + kb + 20);
        v8bf lo, hi;
        lo[0]=(__bf16)a0.x; lo[1]=(__bf16)a0.y; lo[2]=(__bf16)a0.z; lo[3]=(__bf16)a0.w;
        lo[4]=(__bf16)a1.x; lo[5]=(__bf16)a1.y; lo[6]=(__bf16)a1.z; lo[7]=(__bf16)a1.w;
        hi[0]=(__bf16)a2.x; hi[1]=(__bf16)a2.y; hi[2]=(__bf16)a2.z; hi[3]=(__bf16)a2.w;
        hi[4]=(__bf16)a3.x; hi[5]=(__bf16)a3.y; hi[6]=(__bf16)a3.z; hi[7]=(__bf16)a3.w;
        v8bf* sp = (v8bf*)(As + (size_t)(kt * 32 + lane) * 16);
        sp[0] = lo; sp[1] = hi;
    }
    __syncthreads();

    const int wave = t >> 5, lane = t & 31;
    const int half = wave >> 2;          // 0 -> Wz cols, 1 -> Wc cols
    const int tcg  = wave & 3;           // 4 col tiles per wave
    const __bf16* P    = half ? Pc : Pz;
    const float*  bias = half ? bc : bz;
    const int m = lane & 15, hl = lane >> 4;

    v8f acc[4] = {};
#pragma unroll
    for (int ks = 0; ks < 4; ++ks) {
        int kt = half * 4 + ks;
        const v8bf* ap = (const v8bf*)(As + (size_t)(kt * 32 + lane) * 16);
        v16bf A = assemble16(ap[0], ap[1]);
#pragma unroll
        for (int j = 0; j < 4; ++j) {
            int nt = tcg * 4 + j;
            const v8bf* bp = (const v8bf*)(P + ((size_t)(ks * 16 + nt) * 32 + lane) * 16);
            v16bf B = assemble16(bp[0], bp[1]);
            acc[j] = __builtin_amdgcn_wmma_f32_16x16x32_bf16(false, A, false, B,
                                                             (short)0, acc[j], false, false);
        }
    }
    // epilogue: dis row-scale, bias, tanh, bf16 store
    const int rbase = r0 + hl * 8;
    float dv[8];
#pragma unroll
    for (int i = 0; i < 8; ++i) dv[i] = dis[rbase + i];
#pragma unroll
    for (int j = 0; j < 4; ++j) {
        int nt = tcg * 4 + j;
        int col = half * 256 + nt * 16 + m;
        float bj = bias[nt * 16 + m];
#pragma unroll
        for (int i = 0; i < 8; ++i) {
            float val = tanhf(dv[i] * acc[j][i] + bj);
            h[(size_t)(rbase + i) * F_H2 + col] = (__bf16)val;
        }
    }
}

// ------------- GEMM2: h3s = dis ⊙ (h @ Wo); also init d_out (self loop) -------------
// Block of 8 waves = 8 row tiles. Full packed Wo panel (128 KB bf16) staged in dynamic
// LDS once per block; each wave computes all 8 column tiles (8 accumulators).
__global__ __launch_bounds__(256) void k_gemm2(const __bf16* __restrict__ h,
                                               const float* __restrict__ dis,
                                               const __bf16* __restrict__ Po,
                                               float* __restrict__ h3s,
                                               float* __restrict__ outinit) {
    extern __shared__ __bf16 Bs[];       // 512*128 halves = 128 KB
    const int t = threadIdx.x;
    {   // cooperative Wo staging: 8192 x 16B chunks
        const v8bf* gp = (const v8bf*)Po;
        v8bf* sp = (v8bf*)Bs;
        for (int i = t; i < 8192; i += 256) sp[i] = gp[i];
    }
    __syncthreads();

    const int wave = t >> 5, lane = t & 31;
    const int tr = blockIdx.x * 8 + wave;        // wave-uniform
    if (tr >= ROW_TILES) return;
    const int m = lane & 15, hl = lane >> 4;
    const int r = tr * 16 + m;
    const __bf16* arow = h + (size_t)r * F_H2;

    v8f acc[8] = {};
    for (int ks = 0; ks < 16; ++ks) {
        int kb = ks * 32 + hl * 8;
        v8bf alo = *(const v8bf*)(arow + kb);
        v8bf ahi = *(const v8bf*)(arow + kb + 16);
        v16bf A = assemble16(alo, ahi);
#pragma unroll
        for (int j = 0; j < 8; ++j) {
            const v8bf* bp = (const v8bf*)(Bs + ((size_t)(ks * 8 + j) * 32 + lane) * 16);
            v16bf B = assemble16(bp[0], bp[1]);
            acc[j] = __builtin_amdgcn_wmma_f32_16x16x32_bf16(false, A, false, B,
                                                             (short)0, acc[j], false, false);
        }
    }
    const int rbase = tr * 16 + hl * 8;
    float dv[8];
#pragma unroll
    for (int i = 0; i < 8; ++i) dv[i] = dis[rbase + i];
#pragma unroll
    for (int j = 0; j < 8; ++j) {
        int col = j * 16 + m;
#pragma unroll
        for (int i = 0; i < 8; ++i) {
            float v = dv[i] * acc[j][i];
            h3s[(size_t)(rbase + i) * F_OUT + col] = v;
            outinit[(size_t)(rbase + i) * F_OUT + col] = v;
        }
    }
}

// ------------- edge scatter, stage C (128 f32 / edge) -------------
__global__ void k_scatter3(const long long* __restrict__ ei,
                           const float* __restrict__ h3s, float* __restrict__ out) {
    int tid = blockIdx.x * blockDim.x + threadIdx.x;   // E*32 threads
    int e = tid >> 5, c = (tid & 31) << 2;
    if (e >= N_EDGES) return;
    int src = (int)ei[e];
    int dst = (int)ei[N_EDGES + e];
    float4 v = *(const float4*)(h3s + (size_t)src * F_OUT + c);
    float* o = out + (size_t)dst * F_OUT + c;
    unsafeAtomicAdd(o + 0, v.x);
    unsafeAtomicAdd(o + 1, v.y);
    unsafeAtomicAdd(o + 2, v.z);
    unsafeAtomicAdd(o + 3, v.w);
}

// ------------- final: out = dis ⊙ out + b_o -------------
__global__ void k_final(float* __restrict__ out, const float* __restrict__ dis,
                        const float* __restrict__ bo) {
    int tid = blockIdx.x * blockDim.x + threadIdx.x;   // N*32 threads
    int n = tid >> 5, c = (tid & 31) << 2;
    if (n >= N_NODES) return;
    float d = dis[n];
    float4* o = (float4*)(out + (size_t)n * F_OUT + c);
    float4 v = *o;
    float4 b = *(const float4*)(bo + c);
    v.x = d * v.x + b.x;
    v.y = d * v.y + b.y;
    v.z = d * v.z + b.z;
    v.w = d * v.w + b.w;
    *o = v;
}

extern "C" void kernel_launch(void* const* d_in, const int* in_sizes, int n_in,
                              void* d_out, int out_size, void* d_ws, size_t ws_size,
                              hipStream_t stream) {
    const float*     latent = (const float*)d_in[0];
    const float*     cond   = (const float*)d_in[1];
    const long long* ei     = (const long long*)d_in[2];
    const float*     Wz     = (const float*)d_in[3];
    const float*     bz     = (const float*)d_in[4];
    const float*     Wc     = (const float*)d_in[5];
    const float*     bc     = (const float*)d_in[6];
    const float*     Wo     = (const float*)d_in[7];
    const float*     bo     = (const float*)d_in[8];

    char*   ws  = (char*)d_ws;
    float*  dis = (float*)(ws + OFF_DIS);          // deg, then dis in-place
    float*  xs  = (float*)(ws + OFF_XS);
    float*  agg = (float*)(ws + OFF_AGG);
    __bf16* Pz  = (__bf16*)(ws + OFF_PACK);
    __bf16* Pc  = Pz + 128 * 256;
    __bf16* Po  = Pc + 128 * 256;
    __bf16* h   = (__bf16*)(ws + OFF_XS);          // reuse xs region (bf16 N x 512)
    float*  h3s = (float*)(ws + OFF_AGG);          // reuse agg region (f32 N x 128)
    float*  out = (float*)d_out;

    const dim3 B(256);
    k_init_deg  <<<(N_NODES + 255) / 256, B, 0, stream>>>(dis);
    k_count_deg <<<(N_EDGES + 255) / 256, B, 0, stream>>>(ei, dis);
    k_dis       <<<(N_NODES + 255) / 256, B, 0, stream>>>(dis);

    k_pack<<<(4 * 16 * 32 + 255) / 256, B, 0, stream>>>(Wz, Pz, 4, 16, 256);
    k_pack<<<(4 * 16 * 32 + 255) / 256, B, 0, stream>>>(Wc, Pc, 4, 16, 256);
    k_pack<<<(16 * 8 * 32 + 255) / 256, B, 0, stream>>>(Wo, Po, 16, 8, 128);

    k_scale_init<<<(N_NODES * 64 + 255) / 256, B, 0, stream>>>(latent, cond, dis,
                                                               (float4*)xs, (float4*)agg);
    k_scatter1<<<N_EDGES / 4, B, 0, stream>>>(ei, xs, agg);         // E*64 / 256

    k_gemm1<<<ROW_TILES, B, 0, stream>>>(agg, dis, Pz, Pc, bz, bc, h);
    k_gemm2<<<(ROW_TILES + 7) / 8, B, 512 * 128 * sizeof(__bf16), stream>>>(h, dis, Po,
                                                                            h3s, out);

    k_scatter3<<<N_EDGES / 8, B, 0, stream>>>(ei, h3s, out);        // E*32 / 256
    k_final<<<(N_NODES * 32 + 255) / 256, B, 0, stream>>>(out, dis, bo);
}
